// DeformableConv2d_42958262895028
// MI455X (gfx1250) — matrix operand order repaired and checked by reading, not compiled, not run on hardware
//
#include <hip/hip_runtime.h>
#include <hip/hip_bf16.h>

#define BATCH 8
#define CIN   256
#define COUT  256
#define HH    64
#define WW    64
#define HWSZ  4096            // 64*64
#define KK    9
#define RTOT  2304            // KK*CIN
#define NCHUNK 72             // KK * (CIN/32)

typedef __attribute__((ext_vector_type(16))) __bf16 v16bf;
typedef __attribute__((ext_vector_type(8)))  float  v8f;
typedef __attribute__((ext_vector_type(2)))  __bf16 bf16x2;

__device__ __forceinline__ float bf2f(unsigned short u) {
    unsigned int x = ((unsigned int)u) << 16;
    return __builtin_bit_cast(float, x);
}
__device__ __forceinline__ unsigned short f2bf(float f) {
    unsigned int x = __builtin_bit_cast(unsigned int, f);
    unsigned int r = (x + 0x7FFFu + ((x >> 16) & 1u)) >> 16;   // RNE
    return (unsigned short)r;
}

// Async global -> LDS copy (ISA 15.18.3 GLOBAL_LOAD_ASYNC_TO_LDS_B128, GV mode).
__device__ __forceinline__ void async_copy_b128(const void* gsrc, void* ldst) {
    unsigned int       lds_off = (unsigned int)(unsigned long long)(uintptr_t)ldst;
    unsigned long long gaddr   = (unsigned long long)(uintptr_t)gsrc;
    asm volatile("global_load_async_to_lds_b128 %0, %1, off"
                 :: "v"(lds_off), "v"(gaddr) : "memory");
}
__device__ __forceinline__ void async_wait_all() {
    asm volatile("s_wait_asynccnt 0x0" ::: "memory");
}

// ---------------- pre-pass: x NCHW f32 -> NHWC bf16 ----------------
__global__ __launch_bounds__(256) void pack_x_kernel(const float* __restrict__ x,
                                                     unsigned short* __restrict__ xT) {
    int i = blockIdx.x * 256 + threadIdx.x;           // i indexes xT (NHWC)
    int c    = i & (CIN - 1);
    int rest = i >> 8;
    int hw   = rest & (HWSZ - 1);
    int b    = rest >> 12;
    xT[i] = f2bf(x[((size_t)(b * CIN + c)) * HWSZ + hw]);
}

// ---------------- pre-pass: repack weights ----------------
// wBt[o][kpos*256+c] bf16 ; offWT[r][18] f32 with r = kpos*256+c
__global__ __launch_bounds__(256) void pack_w_kernel(const float* __restrict__ conv_w,
                                                     const float* __restrict__ offset_w,
                                                     unsigned short* __restrict__ wBt,
                                                     float* __restrict__ offWT) {
    int i = blockIdx.x * 256 + threadIdx.x;
    if (i < COUT * RTOT) {
        int r = i % RTOT;
        int o = i / RTOT;
        int kpos = r >> 8;
        int c    = r & (CIN - 1);
        wBt[i] = f2bf(conv_w[((size_t)(o * CIN + c)) * KK + kpos]);
    }
    if (i < RTOT * 18) {
        int ko = i % 18;
        int r  = i / 18;
        int kpos = r >> 8;
        int c    = r & (CIN - 1);
        offWT[i] = offset_w[((size_t)(ko * CIN + c)) * KK + kpos];
    }
}

// ---------------- offset branch: plain 3x3 conv -> off[b][ko][hw] f32 ----------------
__global__ __launch_bounds__(256) void offset_conv_kernel(const unsigned short* __restrict__ xT,
                                                          const float* __restrict__ offWT,
                                                          const float* __restrict__ offset_b,
                                                          float* __restrict__ off) {
    int t   = blockIdx.x * 256 + threadIdx.x;
    int ko  = t & 31;
    int pix = t >> 5;
    if (ko >= 18) return;
    int b  = pix >> 12;
    int hw = pix & (HWSZ - 1);
    int h  = hw >> 6;
    int w  = hw & 63;

    float acc = 0.f;
    #pragma unroll
    for (int kpos = 0; kpos < KK; ++kpos) {
        int y = h - 1 + kpos / 3;
        int x = w - 1 + kpos % 3;
        if (y < 0 || y >= HH || x < 0 || x >= WW) continue;
        const unsigned short* src = xT + (((size_t)(b * HH + y) * WW + x) << 8);
        const float* wr = offWT + (size_t)kpos * CIN * 18 + ko;
        #pragma unroll 4
        for (int c = 0; c < CIN; ++c)
            acc += bf2f(src[c]) * wr[(size_t)c * 18];
    }
    off[((size_t)b * 18 + ko) * HWSZ + hw] = acc + offset_b[ko];
}

// ---------------- fragment loaders (contiguous k = 16*(lane/16) + s, same in A and B) ----------------
__device__ __forceinline__ v16bf load_frag_a(const unsigned short* As, int m0, int lane) {
    union { v16bf v; unsigned int u[8]; } f;
    int row   = m0 + (lane & 15);
    int khalf = lane >> 4;
    const unsigned int* p = (const unsigned int*)(As + row * 32);
    #pragma unroll
    for (int v = 0; v < 8; ++v) {
        int pairIdx = khalf * 8 + v;   // contiguous K pairs, matches B
        f.u[v] = p[pairIdx];
    }
    return f.v;
}
__device__ __forceinline__ v16bf load_frag_b(const unsigned short* Bs, int n0, int lane) {
    union { v16bf v; unsigned int u[8]; } f;
    int col   = n0 + (lane & 15);
    int khalf = lane >> 4;                                     // lanes 16-31 -> K=16..31
    const unsigned int* p = (const unsigned int*)(Bs + col * 32);
    #pragma unroll
    for (int v = 0; v < 8; ++v)
        f.u[v] = p[khalf * 8 + v];
    return f.v;
}

// ---------------- fused gather + implicit GEMM main kernel ----------------
// grid: 32768/128 blocks; block: 256 threads = 8 waves.
// Tile: 128 pixels x 256 ochans; wave w owns rows [16w,16w+16) and all 16 n-tiles.
// Double-buffered LDS, one barrier per 32-wide k-chunk; async B copy + bilinear
// gather for chunk t+1 overlap the 16 WMMAs of chunk t.
__global__ __launch_bounds__(256) void deform_main_kernel(const unsigned short* __restrict__ xT,
                                                          const unsigned short* __restrict__ wBt,
                                                          const float* __restrict__ off,
                                                          float* __restrict__ out) {
    __shared__ __align__(16) unsigned short As[2][128 * 32];   // 2 x 8 KB
    __shared__ __align__(16) unsigned short Bs[2][256 * 32];   // 2 x 16 KB

    const int tid   = threadIdx.x;
    const int lane  = tid & 31;
    const int wave  = tid >> 5;
    const int pixStart = blockIdx.x * 128;     // 32 blocks per image -> same b per block
    const int b   = pixStart >> 12;
    const int hw0 = pixStart & (HWSZ - 1);

    const int p_loc   = tid >> 1;              // 2 threads per pixel
    const int halfsel = tid & 1;               // which 16-channel half
    const int hw_my   = hw0 + p_loc;
    const int h_my    = hw_my >> 6;
    const int w_my    = hw_my & 63;

    int    baseR[4];
    bf16x2 wR[4];

    // recompute per-pixel sampling state for a given kernel tap (registers only)
    auto compute_corners = [&](int kpos) {
        float offy = off[((size_t)b * 18 + 2 * kpos    ) * HWSZ + hw_my];
        float offx = off[((size_t)b * 18 + 2 * kpos + 1) * HWSZ + hw_my];
        float py = (float)(h_my - 1 + kpos / 3) + offy;
        float px = (float)(w_my - 1 + kpos % 3) + offx;
        float fy = floorf(py), fx = floorf(px);
        int   y0 = (int)fy,    x0 = (int)fx;
        float wy1 = py - fy, wx1 = px - fx;
        float wy0 = 1.f - wy1, wx0 = 1.f - wx1;
        int   ys[4]  = {y0, y0, y0 + 1, y0 + 1};
        int   xs[4]  = {x0, x0 + 1, x0, x0 + 1};
        float wsc[4] = {wy0 * wx0, wy0 * wx1, wy1 * wx0, wy1 * wx1};
        #pragma unroll
        for (int i = 0; i < 4; ++i) {
            bool valid = (ys[i] >= 0) & (ys[i] < HH) & (xs[i] >= 0) & (xs[i] < WW);
            baseR[i] = valid ? (((b * HH + ys[i]) * WW + xs[i]) << 8) : 0;
            __bf16 wb = (__bf16)(valid ? wsc[i] : 0.f);
            wR[i] = (bf16x2){wb, wb};
        }
    };

    // fill buffer `buf` for flat chunk t (kpos = t>>3, cc = t&7)
    auto fill_tile = [&](int t, int buf) {
        const int kpos = t >> 3;
        const int cc   = t & 7;
        const int r0   = kpos * CIN + cc * 32;
        // B tile: [256 ochan][32 k]; one 64B row per thread via async copies
        {
            const unsigned short* gsrc = wBt + (size_t)tid * RTOT + r0;
            unsigned short*       ldst = Bs[buf] + tid * 32;
            async_copy_b128(gsrc,      ldst);
            async_copy_b128(gsrc + 8,  ldst + 8);
            async_copy_b128(gsrc + 16, ldst + 16);
            async_copy_b128(gsrc + 24, ldst + 24);
        }
        // A tile: branchless bilinear gather, packed-bf16 fma
        {
            int c0 = cc * 32 + halfsel * 16;
            bf16x2 accv[8];
            #pragma unroll
            for (int k = 0; k < 8; ++k) accv[k] = (bf16x2){(__bf16)0.f, (__bf16)0.f};
            #pragma unroll
            for (int i = 0; i < 4; ++i) {
                const unsigned short* src = xT + (size_t)baseR[i] + c0;
                uint4 q0 = *(const uint4*)(src);
                uint4 q1 = *(const uint4*)(src + 8);
                const unsigned int qa[8] = {q0.x, q0.y, q0.z, q0.w, q1.x, q1.y, q1.z, q1.w};
                #pragma unroll
                for (int k = 0; k < 8; ++k) {
                    bf16x2 d = __builtin_bit_cast(bf16x2, qa[k]);
                    accv[k] = __builtin_elementwise_fma(d, wR[i], accv[k]);
                }
            }
            unsigned int packed[8];
            #pragma unroll
            for (int k = 0; k < 8; ++k) packed[k] = __builtin_bit_cast(unsigned int, accv[k]);
            uint4* dst = (uint4*)(As[buf] + p_loc * 32 + halfsel * 16);
            dst[0] = make_uint4(packed[0], packed[1], packed[2], packed[3]);
            dst[1] = make_uint4(packed[4], packed[5], packed[6], packed[7]);
        }
    };

    v8f acc[16] = {};

    // ---- pipeline prologue: chunk 0 into buffer 0 ----
    compute_corners(0);
    fill_tile(0, 0);
    async_wait_all();
    __syncthreads();

    int cur = 0;
    for (int t = 0; t < NCHUNK; ++t) {
        // stage chunk t+1 into the other buffer (overlaps WMMAs below)
        if (t + 1 < NCHUNK) {
            if (((t + 1) & 7) == 0) compute_corners((t + 1) >> 3);
            fill_tile(t + 1, cur ^ 1);
        }
        // 16 WMMAs on chunk t: one A frag per wave, swept across all n-tiles
        v16bf afrag = load_frag_a(As[cur], wave * 16, lane);
        #pragma unroll
        for (int n = 0; n < 16; ++n) {
            v16bf bfrag = load_frag_b(Bs[cur], n * 16, lane);
            acc[n] = __builtin_amdgcn_wmma_f32_16x16x32_bf16(
                false, afrag, false, bfrag, (short)0, acc[n], false, false);
        }
        async_wait_all();      // chunk t+1 B tile resident
        __syncthreads();       // chunk t+1 A tile visible; buffers swappable
        cur ^= 1;
    }

    // ---- store D tiles: lane = N (ochan), VGPR index = consecutive pixels ----
    const int khalf = lane >> 4;
    const int col   = lane & 15;
    #pragma unroll
    for (int n = 0; n < 16; ++n) {
        int ochan = n * 16 + col;
        float* op = out + ((size_t)(b * COUT + ochan)) * HWSZ + hw0 + wave * 16 + khalf * 8;
        #pragma unroll
        for (int r = 0; r < 8; ++r)
            op[r] = acc[n][r];
    }
}

extern "C" void kernel_launch(void* const* d_in, const int* in_sizes, int n_in,
                              void* d_out, int out_size, void* d_ws, size_t ws_size,
                              hipStream_t stream) {
    const float* x        = (const float*)d_in[0];
    const float* offset_w = (const float*)d_in[1];
    const float* offset_b = (const float*)d_in[2];
    const float* conv_w   = (const float*)d_in[3];
    float* out = (float*)d_out;

    char* ws = (char*)d_ws;
    unsigned short* xT    = (unsigned short*)(ws);                       // 16 MB
    unsigned short* wBt   = (unsigned short*)(ws + 16777216);            // 1.125 MB
    float*          offWT = (float*)(ws + 16777216 + 1179648);           // 162 KB
    float*          off   = (float*)(ws + 16777216 + 1179648 + 165888);  // 2.25 MB

    // 1) x -> NHWC bf16
    pack_x_kernel<<<(BATCH * HWSZ * CIN) / 256, 256, 0, stream>>>(x, xT);
    // 2) weight repack
    pack_w_kernel<<<(COUT * RTOT + 255) / 256, 256, 0, stream>>>(conv_w, offset_w, wBt, offWT);
    // 3) offset branch conv
    offset_conv_kernel<<<(BATCH * HWSZ * 32) / 256, 256, 0, stream>>>(xT, offWT, offset_b, off);
    // 4) fused gather + WMMA implicit GEMM (full 256-channel N per block)
    deform_main_kernel<<<dim3(BATCH * HWSZ / 128), 256, 0, stream>>>(xT, wBt, off, out);
}
